// AttentionHead_32358283608292
// MI455X (gfx1250) — matrix operand order repaired and checked by reading, not compiled, not run on hardware
//
#include <hip/hip_runtime.h>

// ---------------------------------------------------------------------------
// Causal single-head attention for MI455X (gfx1250, wave32, WMMA).
//  Pass 0:    convert Wq/Wk/Wv (fp32) -> bf16 once (removes all B-fragment
//             conversion VALU from the projection inner loop).
//  Pass 1..3: q/k/v projections  y = x @ W^T + b  via v_wmma_f32_16x16x32_bf16
//             (x converted to bf16 A-frags on the fly, fp32 accumulation;
//              v stored transposed (D,S) so P@V B-frags are contiguous).
//  Pass 4:    flash attention (online softmax) entirely on bf16 WMMA.
// Workspace: 3*1MB/2 (W bf16) + 3 * 8*2048*128 bf16 (q,k,v^T) ~ 14 MB,
// L2-resident (192 MB L2).
// ---------------------------------------------------------------------------

#define BATCH 8
#define SEQ   2048
#define EMB   2048
#define HDIM  128

typedef __attribute__((ext_vector_type(16))) __bf16 bf16x16;
typedef __attribute__((ext_vector_type(8)))  __bf16 bf16x8;
typedef __attribute__((ext_vector_type(8)))  float  f32x8;

union FragA { bf16x16 v; bf16x8 h[2]; };   // h[0] -> VGPR0-3, h[1] -> VGPR4-7

static __device__ __forceinline__ f32x8 zero8() {
  f32x8 z = {0.f, 0.f, 0.f, 0.f, 0.f, 0.f, 0.f, 0.f};
  return z;
}
static __device__ __forceinline__ bf16x8 cvt8(f32x8 f) {
  return __builtin_convertvector(f, bf16x8);
}
static __device__ __forceinline__ f32x8 wmma_bf16(bf16x16 a, bf16x16 b, f32x8 c) {
  // (neg_a, A, neg_b, B, c_mod, C, reuse_a, reuse_b)
  return __builtin_amdgcn_wmma_f32_16x16x32_bf16(false, a, false, b, (short)0, c,
                                                 false, false);
}

// ---------------------------------------------------------------------------
// fp32 -> bf16 bulk convert (for the three weight matrices, D*E elements).
// ---------------------------------------------------------------------------
__global__ __launch_bounds__(256) void cvt_w(const float* __restrict__ in,
                                             __bf16* __restrict__ out) {
  const int i = blockIdx.x * blockDim.x + threadIdx.x;  // one f32x8 per thread
  const f32x8 v = *(const f32x8*)(in + (size_t)i * 8);
  *(bf16x8*)(out + (size_t)i * 8) = cvt8(v);
}

// ---------------------------------------------------------------------------
// Projection: out(row, d) = sum_e x(row, e) * W(d, e) + bias(d)
// One wave per 16-row x 64-col tile (4 accumulator tiles) -> 2048 waves per
// projection for occupancy.  A-frag: two 32B fp32 runs of the x row converted
// to bf16.  B-frag: single contiguous 32B bf16 load from pre-converted W
// (K = 16*half..+15 slice of the W row).
// ---------------------------------------------------------------------------
template <bool TRANSPOSED>
__global__ __launch_bounds__(256) void proj_wmma(const float* __restrict__ x,
                                                 const __bf16* __restrict__ Wb,
                                                 const float* __restrict__ bias,
                                                 __bf16* __restrict__ out) {
  const int lane  = threadIdx.x & 31;
  const int wv    = threadIdx.x >> 5;
  const int half  = lane >> 4;
  const int col   = lane & 15;
  const int wid   = blockIdx.x * 8 + wv;
  const int row0  = (wid >> 1) * 16;       // 16-row tile over flattened (B*S)
  const int d0    = (wid & 1) * 64;        // which half of the 128 output cols

  f32x8 acc[4];
#pragma unroll
  for (int t = 0; t < 4; ++t) acc[t] = zero8();

  const float* arow = x + (size_t)(row0 + col) * EMB;  // A row = lane&15

  for (int e0 = 0; e0 < EMB; e0 += 32) {
    FragA a;
    a.h[0] = cvt8(*(const f32x8*)(arow + e0 + 16 * half));
    a.h[1] = cvt8(*(const f32x8*)(arow + e0 + 16 * half + 8));
#pragma unroll
    for (int t = 0; t < 4; ++t) {
      const __bf16* wrow =
          Wb + (size_t)(d0 + t * 16 + col) * EMB + e0 + 16 * half;
      bf16x16 b = *(const bf16x16*)wrow;   // one 32B load, no cvt
      acc[t] = wmma_bf16(a.v, b, acc[t]);
    }
  }

#pragma unroll
  for (int t = 0; t < 4; ++t) {
    const int d = d0 + t * 16 + col;
    const float bv = bias[d];
#pragma unroll
    for (int r = 0; r < 8; ++r) {
      const int row = row0 + r + 8 * half;   // C/D: VGPR r -> rows r, r+8
      const float val = acc[t][r] + bv;
      if (!TRANSPOSED) {
        out[(size_t)row * HDIM + d] = (__bf16)val;
      } else {  // v^T layout: (batch, D, S)
        const int bb = row / SEQ, ss = row % SEQ;
        out[((size_t)bb * HDIM + d) * SEQ + ss] = (__bf16)val;
      }
    }
  }
}

// ---------------------------------------------------------------------------
// Flash attention: one wave = 16 query rows, keys in blocks of 32.
// scores (2x 16x16 C tiles) -> mask/scale -> online softmax (shfl_xor row
// reductions within each 16-lane half; stats live per-lane matching the C/D
// row layout) -> P staged via LDS (C layout -> A layout) -> P@V accumulated
// into a 16x128 fp32 O.  Softmax VALU co-executes with the XDL WMMA pipe.
// ---------------------------------------------------------------------------
__global__ __launch_bounds__(256) void attn_wmma(const __bf16* __restrict__ qb,
                                                 const __bf16* __restrict__ kb,
                                                 const __bf16* __restrict__ vt,
                                                 float* __restrict__ out) {
  __shared__ __align__(16) __bf16 pshm[8][16][32];  // per-wave P staging, 8 KB

  const int lane = threadIdx.x & 31;
  const int wv   = threadIdx.x >> 5;
  const int half = lane >> 4;
  const int col  = lane & 15;
  const int tile = blockIdx.x * 8 + wv;          // 0..1023
  const int bidx = tile / (SEQ / 16);
  const int q0   = (tile % (SEQ / 16)) * 16;

  const __bf16* kbase = kb + (size_t)bidx * SEQ * HDIM;
  const __bf16* vbase = vt + (size_t)bidx * HDIM * SEQ;

  // Q A-fragments: 16 rows x 128 d, four 32-wide chunks.
  FragA qa[4];
  const __bf16* qrow = qb + ((size_t)bidx * SEQ + q0 + col) * HDIM;
#pragma unroll
  for (int c = 0; c < 4; ++c) {
    qa[c].h[0] = *(const bf16x8*)(qrow + 32 * c + 16 * half);
    qa[c].h[1] = *(const bf16x8*)(qrow + 32 * c + 16 * half + 8);
  }

  f32x8 o[8];
#pragma unroll
  for (int t = 0; t < 8; ++t) o[t] = zero8();
  float mrow[8], lrow[8];
#pragma unroll
  for (int r = 0; r < 8; ++r) { mrow[r] = -__builtin_inff(); lrow[r] = 0.f; }

  const float rscale  = 0.022097086912079608f;  // 1/sqrt(SEQ) (ref scales by seq)
  const float NEG_INF = -__builtin_inff();
  const int   kend    = q0 + 16;                // causal: keys 0 .. q0+15

  for (int k0 = 0; k0 < kend; k0 += 32) {
    // speculative prefetch of next key/value blocks (global_prefetch_b8)
    __builtin_prefetch(kbase + (size_t)(k0 + 32 + col) * HDIM, 0, 1);
    __builtin_prefetch(vbase + (size_t)col * SEQ + k0 + 32, 0, 1);

    // ---- scores: two 16x16 tiles (keys k0..k0+15 and k0+16..k0+31) ----
    f32x8 s0 = zero8(), s1 = zero8();
    const __bf16* kr0 = kbase + (size_t)(k0 + col) * HDIM;
    const __bf16* kr1 = kbase + (size_t)(k0 + 16 + col) * HDIM;
#pragma unroll
    for (int c = 0; c < 4; ++c) {
      bf16x16 b0 = *(const bf16x16*)(kr0 + 32 * c + 16 * half);
      s0 = wmma_bf16(qa[c].v, b0, s0);
      bf16x16 b1 = *(const bf16x16*)(kr1 + 32 * c + 16 * half);
      s1 = wmma_bf16(qa[c].v, b1, s1);
    }

    // ---- causal mask + scale + online softmax ----
    float alpha[8];
#pragma unroll
    for (int r = 0; r < 8; ++r) {
      const int qi = q0 + r + 8 * half;
      float v0 = (k0 + col      <= qi) ? s0[r] * rscale : NEG_INF;
      float v1 = (k0 + 16 + col <= qi) ? s1[r] * rscale : NEG_INF;
      float rm = fmaxf(v0, v1);
      rm = fmaxf(rm, __shfl_xor(rm, 1, 32));
      rm = fmaxf(rm, __shfl_xor(rm, 2, 32));
      rm = fmaxf(rm, __shfl_xor(rm, 4, 32));
      rm = fmaxf(rm, __shfl_xor(rm, 8, 32));   // row max within 16-lane half
      const float mnew = fmaxf(mrow[r], rm);   // finite: key k0 <= qi always
      alpha[r] = __expf(mrow[r] - mnew);
      mrow[r] = mnew;
      const float p0 = __expf(v0 - mnew);
      const float p1 = __expf(v1 - mnew);
      float rs = p0 + p1;
      rs += __shfl_xor(rs, 1, 32);
      rs += __shfl_xor(rs, 2, 32);
      rs += __shfl_xor(rs, 4, 32);
      rs += __shfl_xor(rs, 8, 32);
      lrow[r] = lrow[r] * alpha[r] + rs;
      // stage P (C layout) into LDS so we can reload it in A layout
      pshm[wv][r + 8 * half][col]      = (__bf16)p0;
      pshm[wv][r + 8 * half][16 + col] = (__bf16)p1;
    }
#pragma unroll
    for (int t = 0; t < 8; ++t)
#pragma unroll
      for (int r = 0; r < 8; ++r) o[t][r] *= alpha[r];

    // ---- P @ V : A-frag from LDS, B-frags contiguous from v^T ----
    FragA pa;
    pa.h[0] = *(const bf16x8*)(&pshm[wv][col][16 * half]);
    pa.h[1] = *(const bf16x8*)(&pshm[wv][col][16 * half + 8]);
#pragma unroll
    for (int t = 0; t < 8; ++t) {
      bf16x16 bvf =
          *(const bf16x16*)(vbase + (size_t)(t * 16 + col) * SEQ + k0 + 16 * half);
      o[t] = wmma_bf16(pa.v, bvf, o[t]);
    }
  }

  // ---- normalize + store fp32 output (B, S, D) ----
#pragma unroll
  for (int t = 0; t < 8; ++t) {
    const int d = t * 16 + col;
#pragma unroll
    for (int r = 0; r < 8; ++r) {
      const int row = q0 + r + 8 * half;
      out[((size_t)bidx * SEQ + row) * HDIM + d] = o[t][r] / lrow[r];
    }
  }
}

extern "C" void kernel_launch(void* const* d_in, const int* in_sizes, int n_in,
                              void* d_out, int out_size, void* d_ws, size_t ws_size,
                              hipStream_t stream) {
  const float* x  = (const float*)d_in[0];
  const float* Wq = (const float*)d_in[1];
  const float* bq = (const float*)d_in[2];
  const float* Wk = (const float*)d_in[3];
  const float* bk = (const float*)d_in[4];
  const float* Wv = (const float*)d_in[5];
  const float* bv = (const float*)d_in[6];

  const size_t wsz = (size_t)HDIM * EMB;              // 262144 elems per W
  const size_t qsz = (size_t)BATCH * SEQ * HDIM;      // 2097152 elems

  __bf16* wqb   = (__bf16*)d_ws;                      // bf16 weights
  __bf16* wkb   = wqb + wsz;
  __bf16* wvb   = wkb + wsz;
  __bf16* qbuf  = wvb + wsz;                          // (B*S, 128) bf16
  __bf16* kbuf  = qbuf + qsz;                         // (B*S, 128) bf16
  __bf16* vtbuf = kbuf + qsz;                         // (B, 128, S) bf16

  // Pass 0: weights fp32 -> bf16 (one f32x8 per thread; 262144/8/256 blocks).
  const dim3 cblk(256), cgrd(wsz / 8 / 256);
  cvt_w<<<cgrd, cblk, 0, stream>>>(Wq, wqb);
  cvt_w<<<cgrd, cblk, 0, stream>>>(Wk, wkb);
  cvt_w<<<cgrd, cblk, 0, stream>>>(Wv, wvb);

  // Pass 1..3: projections. 2048 waves each (16x64 tile per wave).
  const dim3 blk(256);                   // 8 wave32 per block
  const dim3 pgrd(BATCH * SEQ * 2 / 128);
  proj_wmma<false><<<pgrd, blk, 0, stream>>>(x, wqb, bq, qbuf);
  proj_wmma<false><<<pgrd, blk, 0, stream>>>(x, wkb, bk, kbuf);
  proj_wmma<true ><<<pgrd, blk, 0, stream>>>(x, wvb, bv, vtbuf);

  // Pass 4: flash attention, 1024 waves (16 query rows per wave).
  const dim3 agrd(BATCH * SEQ / 128);
  attn_wmma<<<agrd, blk, 0, stream>>>(qbuf, kbuf, vtbuf, (float*)d_out);
}
